// GATRegGraphPrediction_61125974557482
// MI455X (gfx1250) — matrix operand order repaired and checked by reading, not compiled, not run on hardware
//
#include <hip/hip_runtime.h>
#include <math.h>

typedef __attribute__((ext_vector_type(16))) _Float16 v16h;
typedef __attribute__((ext_vector_type(8)))  _Float16 v8h;
typedef __attribute__((ext_vector_type(8)))  float    v8f;

#define N_NODES 50000
#define N_EDGES 800000
#define HID 128
#define HEADS 4
#define C_OUT 32
#define NEG_SLOPE 0.2f

// ---------- helpers: order-preserving float <-> uint encoding for atomicMax ----------
static __device__ __forceinline__ unsigned fenc(float f) {
    unsigned u = __float_as_uint(f);
    return (u & 0x80000000u) ? ~u : (u | 0x80000000u);
}
static __device__ __forceinline__ float fdec(unsigned e) {
    unsigned u = (e & 0x80000000u) ? (e ^ 0x80000000u) : ~e;
    return __uint_as_float(u);
}

// ---------- utility kernels ----------
__global__ void fill_u32_kernel(unsigned* __restrict__ p, unsigned v, int n) {
    int i = blockIdx.x * blockDim.x + threadIdx.x;
    if (i < n) p[i] = v;
}

__global__ void cvt_f32_to_f16_kernel(const float* __restrict__ in,
                                      _Float16* __restrict__ out, int n) {
    int i = blockIdx.x * blockDim.x + threadIdx.x;
    if (i < n) out[i] = (_Float16)in[i];
}

// W [K x M] row-major f32 -> Wt [M x K] row-major f16 (i.e., column-major W)
__global__ void transpose2_f16_kernel(const float* __restrict__ Wl,
                                      const float* __restrict__ Wr,
                                      _Float16* __restrict__ Wtl,
                                      _Float16* __restrict__ Wtr) {
    int i = blockIdx.x * blockDim.x + threadIdx.x;
    if (i >= HID * HID) return;
    int k = i / HID, n = i % HID;
    Wtl[n * HID + k] = (_Float16)Wl[k * HID + n];
    Wtr[n * HID + k] = (_Float16)Wr[k * HID + n];
}

// ---------- WMMA dual GEMM: Yl = A@Bl + bl, Yr = A@Br + br ----------
// A: [nrows x 128] f16 row-major.  Bl/Br: [128 x 128] f16 stored column-major
// (Bt[n*128+k]).  One block = 16 rows x 128 cols; 8 waves, wave w owns column
// tile w.  K=128 -> 4 x v_wmma_f32_16x16x32_f16 per accumulator.
__global__ __launch_bounds__(256)
void wmma_dual_gemm_bias_kernel(const _Float16* __restrict__ A,
                                const _Float16* __restrict__ Bl,
                                const _Float16* __restrict__ Br,
                                const float* __restrict__ bl,
                                const float* __restrict__ br,
                                float* __restrict__ Yl,
                                float* __restrict__ Yr) {
    const int lane = threadIdx.x & 31;
    const int wave = threadIdx.x >> 5;     // 0..7 -> 16-col tile
    const int l16  = lane & 15;
    const int hf   = lane >> 4;            // lane half (0/1)
    const int row  = blockIdx.x * 16 + l16;  // A row held by this lane
    const int col  = wave * 16 + l16;        // B column held by this lane

    const _Float16* __restrict__ arow  = A  + (size_t)row * HID;
    const _Float16* __restrict__ blcol = Bl + (size_t)col * HID;
    const _Float16* __restrict__ brcol = Br + (size_t)col * HID;

    v8f accl = {};
    v8f accr = {};

#pragma unroll
    for (int kb = 0; kb < HID; kb += 32) {
        // A fragment (16x32 f16): VGPR0-3 = K[hf*16 .. hf*16+7],
        //                         VGPR4-7 = K[hf*16+8 .. hf*16+15]
        union { v16h v; v8h h[2]; } a, fl, fr;
        a.h[0] = *(const v8h*)(arow + kb + hf * 16);
        a.h[1] = *(const v8h*)(arow + kb + hf * 16 + 8);
        // B fragment (32x16 f16): lanes 0-15 K=0..15, lanes 16-31 K=16..31,
        // 2 K-values packed per VGPR -> 16 contiguous f16 per lane.
        fl.h[0] = *(const v8h*)(blcol + kb + hf * 16);
        fl.h[1] = *(const v8h*)(blcol + kb + hf * 16 + 8);
        fr.h[0] = *(const v8h*)(brcol + kb + hf * 16);
        fr.h[1] = *(const v8h*)(brcol + kb + hf * 16 + 8);

        accl = __builtin_amdgcn_wmma_f32_16x16x32_f16(
            false, a.v, false, fl.v, (short)0, accl, false, false);
        accr = __builtin_amdgcn_wmma_f32_16x16x32_f16(
            false, a.v, false, fr.v, (short)0, accr, false, false);
    }

    const float biasl = bl[col];
    const float biasr = br[col];
#pragma unroll
    for (int r = 0; r < 8; ++r) {
        // D layout: VGPR r -> M = r + 8*hf, N = lane%16
        int orow = blockIdx.x * 16 + r + hf * 8;
        Yl[(size_t)orow * HID + col] = accl[r] + biasl;
        Yr[(size_t)orow * HID + col] = accr[r] + biasr;
    }
}

// ---------- edge kernels, layers 1-2 ----------
// logits[e,h] = sum_c att[h,c] * leaky_relu(xl[src,h,c] + xr[dst,h,c]); segment max
__global__ void edge_logits_kernel(const float* __restrict__ xl,
                                   const float* __restrict__ xr,
                                   const int* __restrict__ src,
                                   const int* __restrict__ dst,
                                   const float* __restrict__ att,
                                   float* __restrict__ logits,
                                   unsigned* __restrict__ mmax) {
    int i = blockIdx.x * blockDim.x + threadIdx.x;
    if (i >= N_EDGES * HEADS) return;
    int e = i >> 2, h = i & 3;
    int s = src[e], d = dst[e];
    const float* pl = xl + (size_t)s * HID + h * C_OUT;
    const float* pr = xr + (size_t)d * HID + h * C_OUT;
    const float* pa = att + h * C_OUT;
    float acc = 0.f;
#pragma unroll
    for (int c = 0; c < C_OUT; c += 4) {
        float4 a = *(const float4*)(pl + c);
        float4 b = *(const float4*)(pr + c);
        float4 w = *(const float4*)(pa + c);
        float v0 = a.x + b.x; v0 = v0 > 0.f ? v0 : NEG_SLOPE * v0;
        float v1 = a.y + b.y; v1 = v1 > 0.f ? v1 : NEG_SLOPE * v1;
        float v2 = a.z + b.z; v2 = v2 > 0.f ? v2 : NEG_SLOPE * v2;
        float v3 = a.w + b.w; v3 = v3 > 0.f ? v3 : NEG_SLOPE * v3;
        acc += w.x * v0 + w.y * v1 + w.z * v2 + w.w * v3;
    }
    logits[i] = acc;
    atomicMax(mmax + (size_t)d * HEADS + h, fenc(acc));
}

// alpha = exp(logit - max); denom[dst,h] += alpha; num[dst,h,:] += xl[src,h,:]*alpha
__global__ void edge_agg_kernel(const float* __restrict__ xl,
                                const int* __restrict__ src,
                                const int* __restrict__ dst,
                                const float* __restrict__ logits,
                                const unsigned* __restrict__ mmax,
                                float* __restrict__ denom,
                                float* __restrict__ num) {
    int i = blockIdx.x * blockDim.x + threadIdx.x;
    if (i >= N_EDGES * HEADS) return;
    int e = i >> 2, h = i & 3;
    int s = src[e], d = dst[e];
    float m = fdec(mmax[(size_t)d * HEADS + h]);
    float alpha = expf(logits[i] - m);
    atomicAdd(denom + (size_t)d * HEADS + h, alpha);
    const float* pl = xl + (size_t)s * HID + h * C_OUT;
    float* pn = num + (size_t)d * HID + h * C_OUT;
#pragma unroll
    for (int c = 0; c < C_OUT; ++c)
        atomicAdd(pn + c, pl[c] * alpha);
}

// out = num/(denom+eps) + bias; optional ELU; in place
__global__ void node_finalize_kernel(float* __restrict__ num,
                                     const float* __restrict__ denom,
                                     const float* __restrict__ bias,
                                     int apply_elu) {
    int i = blockIdx.x * blockDim.x + threadIdx.x;
    if (i >= N_NODES * HID) return;
    int nidx = i / HID, ch = i % HID, h = ch / C_OUT;
    float v = num[i] / (denom[(size_t)nidx * HEADS + h] + 1e-16f) + bias[ch];
    if (apply_elu) v = v > 0.f ? v : expf(v) - 1.f;
    num[i] = v;
}

// ---------- layer 3 (128 -> 1, concat=False) ----------
__global__ void gemv3_kernel(const float* __restrict__ X,
                             const float* __restrict__ Wl, const float* __restrict__ bl,
                             const float* __restrict__ Wr, const float* __restrict__ br,
                             float* __restrict__ xl, float* __restrict__ xr) {
    int n = blockIdx.x * blockDim.x + threadIdx.x;
    if (n >= N_NODES) return;
    const float* px = X + (size_t)n * HID;
    float al = 0.f, ar = 0.f;
#pragma unroll 4
    for (int k = 0; k < HID; ++k) {
        float v = px[k];
        al += v * Wl[k];
        ar += v * Wr[k];
    }
    xl[n] = al + bl[0];
    xr[n] = ar + br[0];
}

__global__ void edge3_logit_kernel(const float* __restrict__ xl,
                                   const float* __restrict__ xr,
                                   const int* __restrict__ src,
                                   const int* __restrict__ dst,
                                   const float* __restrict__ att,
                                   float* __restrict__ logits,
                                   unsigned* __restrict__ mmax) {
    int e = blockIdx.x * blockDim.x + threadIdx.x;
    if (e >= N_EDGES) return;
    float v = xl[src[e]] + xr[dst[e]];
    v = v > 0.f ? v : NEG_SLOPE * v;
    float lg = att[0] * v;
    logits[e] = lg;
    atomicMax(mmax + dst[e], fenc(lg));
}

__global__ void edge3_agg_kernel(const float* __restrict__ xl,
                                 const int* __restrict__ src,
                                 const int* __restrict__ dst,
                                 const float* __restrict__ logits,
                                 const unsigned* __restrict__ mmax,
                                 float* __restrict__ denom,
                                 float* __restrict__ num) {
    int e = blockIdx.x * blockDim.x + threadIdx.x;
    if (e >= N_EDGES) return;
    int s = src[e], d = dst[e];
    float alpha = expf(logits[e] - fdec(mmax[d]));
    atomicAdd(denom + d, alpha);
    atomicAdd(num + d, xl[s] * alpha);
}

__global__ void node3_final_kernel(const float* __restrict__ num,
                                   const float* __restrict__ denom,
                                   const float* __restrict__ bias,
                                   float* __restrict__ out) {
    int n = blockIdx.x * blockDim.x + threadIdx.x;
    if (n >= N_NODES) return;
    float v = num[n] / (denom[n] + 1e-16f) + bias[0];
    out[n] = 1.f / (1.f + expf(-v));
}

// ---------- host launch ----------
extern "C" void kernel_launch(void* const* d_in, const int* in_sizes, int n_in,
                              void* d_out, int out_size, void* d_ws, size_t ws_size,
                              hipStream_t stream) {
    const float* x    = (const float*)d_in[0];
    const int*   ei   = (const int*)d_in[1];
    const int*   src  = ei;
    const int*   dst  = ei + N_EDGES;
    const float* Wl1 = (const float*)d_in[2];  const float* bl1 = (const float*)d_in[3];
    const float* Wr1 = (const float*)d_in[4];  const float* br1 = (const float*)d_in[5];
    const float* at1 = (const float*)d_in[6];  const float* bs1 = (const float*)d_in[7];
    const float* Wl2 = (const float*)d_in[8];  const float* bl2 = (const float*)d_in[9];
    const float* Wr2 = (const float*)d_in[10]; const float* br2 = (const float*)d_in[11];
    const float* at2 = (const float*)d_in[12]; const float* bs2 = (const float*)d_in[13];
    const float* Wl3 = (const float*)d_in[14]; const float* bl3 = (const float*)d_in[15];
    const float* Wr3 = (const float*)d_in[16]; const float* br3 = (const float*)d_in[17];
    const float* at3 = (const float*)d_in[18]; const float* bs3 = (const float*)d_in[19];

    char* ws = (char*)d_ws;
    size_t off = 0;
    auto alloc = [&](size_t bytes) -> void* {
        void* p = ws + off;
        off += (bytes + 255) & ~(size_t)255;
        return p;
    };
    float*    xlb   = (float*)alloc((size_t)N_NODES * HID * 4);
    float*    xrb   = (float*)alloc((size_t)N_NODES * HID * 4);
    float*    hnum  = (float*)alloc((size_t)N_NODES * HID * 4);   // num accumulator / layer output
    float*    logit = (float*)alloc((size_t)N_EDGES * HEADS * 4);
    unsigned* mmax  = (unsigned*)alloc((size_t)N_NODES * HEADS * 4);
    float*    denom = (float*)alloc((size_t)N_NODES * HEADS * 4);
    _Float16* Ah    = (_Float16*)alloc((size_t)N_NODES * HID * 2);
    _Float16* Wtl   = (_Float16*)alloc((size_t)HID * HID * 2);
    _Float16* Wtr   = (_Float16*)alloc((size_t)HID * HID * 2);

    const int TB = 256;
    const int nflat  = N_NODES * HID;           // 6.4M
    const int nheads = N_NODES * HEADS;
    const int eheads = N_EDGES * HEADS;

    auto run_layer = [&](const float* input, const float* Wl, const float* bl,
                         const float* Wr, const float* br, const float* att,
                         const float* bias) {
        cvt_f32_to_f16_kernel<<<(nflat + TB - 1) / TB, TB, 0, stream>>>(input, Ah, nflat);
        transpose2_f16_kernel<<<(HID * HID + TB - 1) / TB, TB, 0, stream>>>(Wl, Wr, Wtl, Wtr);
        wmma_dual_gemm_bias_kernel<<<N_NODES / 16, TB, 0, stream>>>(Ah, Wtl, Wtr, bl, br, xlb, xrb);
        fill_u32_kernel<<<(nflat + TB - 1) / TB, TB, 0, stream>>>((unsigned*)hnum, 0u, nflat);
        fill_u32_kernel<<<(nheads + TB - 1) / TB, TB, 0, stream>>>((unsigned*)denom, 0u, nheads);
        fill_u32_kernel<<<(nheads + TB - 1) / TB, TB, 0, stream>>>(mmax, 0u, nheads);
        edge_logits_kernel<<<(eheads + TB - 1) / TB, TB, 0, stream>>>(xlb, xrb, src, dst, att, logit, mmax);
        edge_agg_kernel<<<(eheads + TB - 1) / TB, TB, 0, stream>>>(xlb, src, dst, logit, mmax, denom, hnum);
        node_finalize_kernel<<<(nflat + TB - 1) / TB, TB, 0, stream>>>(hnum, denom, bias, 1);
    };

    run_layer(x,    Wl1, bl1, Wr1, br1, at1, bs1);
    run_layer(hnum, Wl2, bl2, Wr2, br2, at2, bs2);

    // ---- layer 3 (scalar head) ----
    float* xl3  = xlb;            // N floats
    float* xr3  = xrb;            // N floats
    float* lg3  = logit;          // E floats
    float* num3 = (float*)Ah;     // reuse f16 scratch as N floats
    gemv3_kernel<<<(N_NODES + TB - 1) / TB, TB, 0, stream>>>(hnum, Wl3, bl3, Wr3, br3, xl3, xr3);
    fill_u32_kernel<<<(N_NODES + TB - 1) / TB, TB, 0, stream>>>(mmax, 0u, N_NODES);
    fill_u32_kernel<<<(N_NODES + TB - 1) / TB, TB, 0, stream>>>((unsigned*)denom, 0u, N_NODES);
    fill_u32_kernel<<<(N_NODES + TB - 1) / TB, TB, 0, stream>>>((unsigned*)num3, 0u, N_NODES);
    edge3_logit_kernel<<<(N_EDGES + TB - 1) / TB, TB, 0, stream>>>(xl3, xr3, src, dst, at3, lg3, mmax);
    edge3_agg_kernel<<<(N_EDGES + TB - 1) / TB, TB, 0, stream>>>(xl3, src, dst, lg3, mmax, denom, num3);
    node3_final_kernel<<<(N_NODES + TB - 1) / TB, TB, 0, stream>>>(num3, denom, bs3, (float*)d_out);
}
